// mambaDuelingModel_90065464197335
// MI455X (gfx1250) — matrix operand order repaired and checked by reading, not compiled, not run on hardware
//
#include <hip/hip_runtime.h>
#include <hip/hip_bf16.h>
#include <cstdint>
#include <cstddef>

// ---------------- model dims ----------------
#define B_   64
#define T_   300
#define D_   128
#define NL_  4
#define DI_  256
#define DS_  16
#define DTR_ 8
#define DC_  4
#define H1_  512
#define H2_  256
#define NA_  4
#define BT_  (B_ * T_)
#define FLAT_ (T_ * D_)

typedef __attribute__((ext_vector_type(16))) __bf16 v16bf;
typedef __attribute__((ext_vector_type(8)))  float  v8f;

__device__ __forceinline__ float sigmoidf_(float x) { return 1.f / (1.f + __expf(-x)); }
__device__ __forceinline__ float siluf_(float x)    { return x * sigmoidf_(x); }
__device__ __forceinline__ float softplusf_(float x){ return (x > 20.f) ? x : log1pf(__expf(x)); }

// =====================================================================
// Generic GEMM: C[M,N] = A[M,K] @ W[N,K]^T   (N, K, MODE compile-time)
// bf16 WMMA, fp32 accumulate. Block = 128 thr (4 waves); block tile 64x64,
// each wave: 16 rows x 64 cols (4x v_wmma_f32_16x16x32_bf16 per K-step).
// Requires M % 64 == 0 and K % 32 == 0 (true for every call site here).
// MODE 0: C = acc + bias          (gridDim.z == 1)
// MODE 1: C += acc                (gridDim.z == 1, residual add)
// MODE 2: atomicAdd(C, acc)       (split-K: blockIdx.z picks K-slice of KC;
//                                  C must be pre-filled with bias)
// =====================================================================
template<int N, int K, int MODE>
__global__ void k_gemm_wmma(const float* __restrict__ A,
                            const float* __restrict__ W,
                            float* __restrict__ C,
                            const float* __restrict__ bias,
                            int M, int KC)
{
    const int wave = threadIdx.x >> 5;
    const int lane = threadIdx.x & 31;
    const int half = lane & 15;
    const bool hi  = lane >= 16;

    const int m0 = blockIdx.y * 64 + wave * 16;
    const int n0 = blockIdx.x * 64;
    if (m0 >= M) return;                       // uniform per wave -> EXEC stays full

    const int kBegin = blockIdx.z * KC;
    int kLen = K - kBegin; if (kLen > KC) kLen = KC;

    // Fragment source pointers, hoisted; advanced by 32 floats per K-step.
    // A layout per ISA 7.12.2 (16-bit A 16x32): lo lanes K{0..7,16..23}, hi K{8..15,24..31}
    // B layout: same K order as A: lo lanes K{0..7,16..23}, hi K{8..15,24..31} of column n.
    const int abase = hi ? 8 : 0;
    const int bbase = hi ? 8 : 0;
    const float* __restrict__ Ap = A + (size_t)(m0 + half) * K + kBegin + abase;

    bool nvalid[4];
    const float* __restrict__ Wq[4];
    #pragma unroll
    for (int j = 0; j < 4; ++j) {
        const int n = n0 + j * 16 + half;
        nvalid[j] = (n < N);
        Wq[j] = W + (size_t)(nvalid[j] ? n : 0) * K + kBegin + bbase;
    }

    v8f acc[4];
    #pragma unroll
    for (int j = 0; j < 4; ++j)
        #pragma unroll
        for (int r = 0; r < 8; ++r) acc[j][r] = 0.f;

    for (int it = 0; it < kLen; it += 32) {
        if (it + 32 < kLen) __builtin_prefetch(Ap + 32, 0, 1);  // global_prefetch_b8

        v16bf a;
        #pragma unroll
        for (int e = 0; e < 8; ++e) a[e] = (__bf16)Ap[e];          // contiguous 32B
        #pragma unroll
        for (int e = 0; e < 8; ++e) a[8 + e] = (__bf16)Ap[16 + e]; // contiguous 32B
        Ap += 32;

        #pragma unroll
        for (int j = 0; j < 4; ++j) {
            v16bf bf;
            if (nvalid[j]) {
                #pragma unroll
                for (int e = 0; e < 16; ++e) bf[e] = (__bf16)Wq[j][e < 8 ? e : 8 + e];  // 2x contiguous 32B
            } else {
                #pragma unroll
                for (int e = 0; e < 16; ++e) bf[e] = (__bf16)0.0f;
            }
            Wq[j] += 32;
            acc[j] = __builtin_amdgcn_wmma_f32_16x16x32_bf16(
                false, a, false, bf, (short)0, acc[j], false, false);
        }
    }

    // Store per C/D layout: VGPR r -> row m0 + r (+8 for hi lanes), col = half
    #pragma unroll
    for (int j = 0; j < 4; ++j) {
        const int n = n0 + j * 16 + half;
        if (!nvalid[j]) continue;
        const float bv = (MODE == 0 && bias) ? bias[n] : 0.f;
        #pragma unroll
        for (int r = 0; r < 8; ++r) {
            const int m = m0 + r + (hi ? 8 : 0);
            const size_t idx = (size_t)m * N + n;
            if (MODE == 2)      atomicAdd(&C[idx], acc[j][r]);
            else if (MODE == 1) C[idx] += acc[j][r];
            else                C[idx] = acc[j][r] + bv;
        }
    }
}

// Pre-fill C[m, n] = bias[n] (used before split-K atomic GEMMs)
__global__ void k_fill_bias(float* __restrict__ C, const float* __restrict__ bias, int N)
{
    const int m = blockIdx.x;
    for (int n = threadIdx.x; n < N; n += blockDim.x)
        C[(size_t)m * N + n] = bias[n];
}

// =====================================================================
// DAIN (operates on src viewed as (B, D, T); output written back (B,T,D))
// =====================================================================
__global__ void k_dain_avg(const float* __restrict__ src, float* __restrict__ avg)
{
    const int b = blockIdx.x, d = threadIdx.x;
    float s = 0.f;
    for (int t = 0; t < T_; ++t) s += src[((size_t)b * T_ + t) * D_ + d];
    avg[b * D_ + d] = s / (float)T_;
}

// out[b,d] = f( sum_e in[b,e] * W[d,e] (+bias) );  mode 0: id, 1: clamp<=1e-8 -> 1, 2: sigmoid(x+bias)
__global__ void k_dain_mat(const float* __restrict__ in, const float* __restrict__ W,
                           const float* __restrict__ bias, float* __restrict__ out, int mode)
{
    const int b = blockIdx.x, d = threadIdx.x;
    float s = 0.f;
    for (int e = 0; e < D_; ++e) s += in[b * D_ + e] * W[d * D_ + e];
    if (mode == 1)      s = (s <= 1e-8f) ? 1.f : s;
    else if (mode == 2) s = sigmoidf_(s + bias[d]);
    out[b * D_ + d] = s;
}

__global__ void k_dain_std(const float* __restrict__ src, const float* __restrict__ m2,
                           float* __restrict__ stdb)
{
    const int b = blockIdx.x, d = threadIdx.x;
    const float mu = m2[b * D_ + d];
    float s = 0.f;
    for (int t = 0; t < T_; ++t) {
        const float c = src[((size_t)b * T_ + t) * D_ + d] - mu;
        s += c * c;
    }
    stdb[b * D_ + d] = sqrtf(s / (float)T_ + 1e-8f);
}

__global__ void k_dain_mu(const float* __restrict__ src, const float* __restrict__ m2,
                          const float* __restrict__ ad, float* __restrict__ mu)
{
    const int b = blockIdx.x, d = threadIdx.x;
    const float m = m2[b * D_ + d], a = ad[b * D_ + d];
    float s = 0.f;
    for (int t = 0; t < T_; ++t) s += (src[((size_t)b * T_ + t) * D_ + d] - m);
    mu[b * D_ + d] = (s / (float)T_) / a;
}

__global__ void k_dain_write(const float* __restrict__ src, const float* __restrict__ m2,
                             const float* __restrict__ ad, const float* __restrict__ gate,
                             float* __restrict__ x)
{
    const int bt = blockIdx.x, d = threadIdx.x;
    const int b = bt / T_;
    const size_t idx = (size_t)bt * D_ + d;
    x[idx] = (src[idx] - m2[b * D_ + d]) / ad[b * D_ + d] * gate[b * D_ + d];
}

// =====================================================================
// RMSNorm over rows of length D_ (128 threads == one row)
// =====================================================================
__global__ void k_rmsnorm(const float* __restrict__ x, const float* __restrict__ w,
                          float* __restrict__ out)
{
    const int row = blockIdx.x, d = threadIdx.x;
    const float v = x[(size_t)row * D_ + d];
    __shared__ float red[D_];
    red[d] = v * v;
    __syncthreads();
    for (int off = D_ / 2; off > 0; off >>= 1) {
        if (d < off) red[d] += red[d + off];
        __syncthreads();
    }
    const float ms = red[0] / (float)D_;
    out[(size_t)row * D_ + d] = v * w[d] * rsqrtf(ms + 1e-5f);
}

// =====================================================================
// Causal depthwise conv (DC=4) + bias + SiLU.  xz: (BT, 2*DI); xc: (BT, DI)
// =====================================================================
__global__ void k_conv_silu(const float* __restrict__ xz, const float* __restrict__ cW,
                            const float* __restrict__ cb, float* __restrict__ xc)
{
    const int bt = blockIdx.x, i = threadIdx.x;
    const int b = bt / T_, t = bt % T_;
    float acc = cb[i];
    #pragma unroll
    for (int k = 0; k < DC_; ++k) {
        const int tt = t + k - (DC_ - 1);
        if (tt >= 0)
            acc += xz[((size_t)(b * T_ + tt)) * (2 * DI_) + i] * cW[i * DC_ + k];
    }
    xc[(size_t)bt * DI_ + i] = siluf_(acc);
}

// =====================================================================
// dt projection:  dt[bt,i] = softplus( sum_r dbl[bt,r] * dtW[i,r] + dtb[i] )
// =====================================================================
__global__ void k_dtproj(const float* __restrict__ dbl, const float* __restrict__ dtW,
                         const float* __restrict__ dtb, float* __restrict__ dt)
{
    const int bt = blockIdx.x, i = threadIdx.x;
    const float* r = dbl + (size_t)bt * (DTR_ + 2 * DS_);
    float s = dtb[i];
    #pragma unroll
    for (int k = 0; k < DTR_; ++k) s += r[k] * dtW[i * DTR_ + k];
    dt[(size_t)bt * DI_ + i] = softplusf_(s);
}

// =====================================================================
// Selective scan: per (b,i) thread carries 16-state recurrence in registers.
// Per timestep the 32 contiguous B/C floats are staged into LDS with the
// CDNA5 async global->LDS copy (8 lanes x b128, ASYNCcnt + barrier).
// Also fuses +Dp*xc and *silu(zg).
// =====================================================================
__global__ void k_scan(const float* __restrict__ xc, const float* __restrict__ dt,
                       const float* __restrict__ dbl, const float* __restrict__ Alog,
                       const float* __restrict__ Dp, const float* __restrict__ xz,
                       float* __restrict__ y)
{
    const int b = blockIdx.x, i = threadIdx.x;
    __shared__ float sBC[2 * DS_];
    float Ai[DS_], h[DS_];
    #pragma unroll
    for (int s = 0; s < DS_; ++s) { Ai[s] = -__expf(Alog[i * DS_ + s]); h[s] = 0.f; }
    const float dpi = Dp[i];

    for (int t = 0; t < T_; ++t) {
        const size_t bt   = (size_t)b * T_ + t;
        const size_t base = bt * (DTR_ + 2 * DS_) + DTR_;   // B(16) ++ C(16), contiguous
        if (i < 8) {
            const float* g = dbl + base + (size_t)i * 4;
            const unsigned ldsOff = (unsigned)(uintptr_t)(&sBC[i * 4]); // LDS aperture: low 32 bits
            asm volatile("global_load_async_to_lds_b128 %0, %1, off"
                         :: "v"(ldsOff), "v"(g) : "memory");
        }
        asm volatile("s_wait_asynccnt 0x0" ::: "memory");
        __syncthreads();

        const float dtv = dt[bt * DI_ + i];
        const float xv  = xc[bt * DI_ + i];
        float acc = 0.f;
        #pragma unroll
        for (int s = 0; s < DS_; ++s) {
            h[s] = __expf(dtv * Ai[s]) * h[s] + dtv * sBC[s] * xv;
            acc += h[s] * sBC[DS_ + s];
        }
        const float z = xz[bt * (2 * DI_) + DI_ + i];
        y[bt * DI_ + i] = (acc + dpi * xv) * siluf_(z);
        __syncthreads();
    }
}

// =====================================================================
// gMLP gate:  g[bt,i] = h[bt,i] * silu(h[bt, DINT+i])   (h rows of 512)
// =====================================================================
__global__ void k_glu(const float* __restrict__ h, float* __restrict__ g)
{
    const int bt = blockIdx.x, i = threadIdx.x;
    const float* r = h + (size_t)bt * (2 * DI_);
    g[(size_t)bt * DI_ + i] = r[i] * siluf_(r[DI_ + i]);
}

// =====================================================================
// LayerNorm + ReLU over rows of length n (n = 512 or 256), 256 thr/row
// =====================================================================
__global__ void k_lnrelu(const float* __restrict__ in, const float* __restrict__ g,
                         const float* __restrict__ be, float* __restrict__ out, int n)
{
    const int row = blockIdx.x, tid = threadIdx.x;
    const float* p = in + (size_t)row * n;
    __shared__ float s1[256], s2[256];
    float a = 0.f, b2 = 0.f;
    for (int j = tid; j < n; j += 256) { const float v = p[j]; a += v; b2 += v * v; }
    s1[tid] = a; s2[tid] = b2;
    __syncthreads();
    for (int off = 128; off > 0; off >>= 1) {
        if (tid < off) { s1[tid] += s1[tid + off]; s2[tid] += s2[tid + off]; }
        __syncthreads();
    }
    const float mean = s1[0] / (float)n;
    const float var  = s2[0] / (float)n - mean * mean;
    const float rs   = rsqrtf(var + 1e-5f);
    for (int j = tid; j < n; j += 256) {
        const float v = (p[j] - mean) * rs * g[j] + be[j];
        out[(size_t)row * n + j] = v > 0.f ? v : 0.f;
    }
}

// =====================================================================
// Dueling combine: q = val + adv - mean(adv); aux scalar = 0
// =====================================================================
__global__ void k_head_final(const float* __restrict__ h2v, const float* __restrict__ h2a,
                             const float* __restrict__ vW3, const float* __restrict__ vb3,
                             const float* __restrict__ aW3, const float* __restrict__ ab3,
                             float* __restrict__ out)
{
    const int b = threadIdx.x;
    if (b >= B_) return;
    float val = vb3[0];
    for (int k = 0; k < H2_; ++k) val += h2v[(size_t)b * H2_ + k] * vW3[k];
    float adv[NA_], s = 0.f;
    #pragma unroll
    for (int a = 0; a < NA_; ++a) {
        float v = ab3[a];
        for (int k = 0; k < H2_; ++k) v += h2a[(size_t)b * H2_ + k] * aW3[a * H2_ + k];
        adv[a] = v; s += v;
    }
    s *= (1.f / (float)NA_);
    #pragma unroll
    for (int a = 0; a < NA_; ++a) out[b * NA_ + a] = val + adv[a] - s;
    if (b == 0) out[B_ * NA_] = 0.f;   // aux
}

// =====================================================================
// Host-side orchestration
// =====================================================================
extern "C" void kernel_launch(void* const* d_in, const int* in_sizes, int n_in,
                              void* d_out, int out_size, void* d_ws, size_t ws_size,
                              hipStream_t stream)
{
    (void)in_sizes; (void)n_in; (void)out_size; (void)ws_size;

    const float* src  = (const float*)d_in[0];
    const float* dmW  = (const float*)d_in[1];
    const float* dsW  = (const float*)d_in[2];
    const float* dgW  = (const float*)d_in[3];
    const float* dgb  = (const float*)d_in[4];
    const float* n1w  = (const float*)d_in[5];
    const float* ipW  = (const float*)d_in[6];
    const float* cW   = (const float*)d_in[7];
    const float* cb   = (const float*)d_in[8];
    const float* xpW  = (const float*)d_in[9];
    const float* dtW  = (const float*)d_in[10];
    const float* dtb  = (const float*)d_in[11];
    const float* Alog = (const float*)d_in[12];
    const float* Dp   = (const float*)d_in[13];
    const float* opW  = (const float*)d_in[14];
    const float* n2w  = (const float*)d_in[15];
    const float* f1W  = (const float*)d_in[16];
    const float* f2W  = (const float*)d_in[17];
    const float* nfw  = (const float*)d_in[18];
    const float* vW1  = (const float*)d_in[19];
    const float* vb1  = (const float*)d_in[20];
    const float* vln1w= (const float*)d_in[21];
    const float* vln1b= (const float*)d_in[22];
    const float* vW2  = (const float*)d_in[23];
    const float* vb2  = (const float*)d_in[24];
    const float* vln2w= (const float*)d_in[25];
    const float* vln2b= (const float*)d_in[26];
    const float* vW3  = (const float*)d_in[27];
    const float* vb3  = (const float*)d_in[28];
    const float* aW1  = (const float*)d_in[29];
    const float* ab1  = (const float*)d_in[30];
    const float* aln1w= (const float*)d_in[31];
    const float* aln1b= (const float*)d_in[32];
    const float* aW2  = (const float*)d_in[33];
    const float* ab2  = (const float*)d_in[34];
    const float* aln2w= (const float*)d_in[35];
    const float* aln2b= (const float*)d_in[36];
    const float* aW3  = (const float*)d_in[37];
    const float* ab3  = (const float*)d_in[38];

    float* ws = (float*)d_ws;
    size_t off = 0;
    auto alloc = [&](size_t nfloats) { float* p = ws + off; off += nfloats; return p; };

    float* x    = alloc((size_t)BT_ * D_);          // residual stream
    float* xn   = alloc((size_t)BT_ * D_);          // normed / final flat
    float* xz   = alloc((size_t)BT_ * 2 * DI_);     // ip proj / gmlp hidden
    float* xc   = alloc((size_t)BT_ * DI_);         // conv out / glu out
    float* dbl  = alloc((size_t)BT_ * (DTR_ + 2 * DS_));
    float* dtv  = alloc((size_t)BT_ * DI_);
    float* yb   = alloc((size_t)BT_ * DI_);
    float* avg  = alloc((size_t)B_ * D_);
    float* m2   = alloc((size_t)B_ * D_);
    float* stdb = alloc((size_t)B_ * D_);
    float* adb  = alloc((size_t)B_ * D_);
    float* mub  = alloc((size_t)B_ * D_);
    float* gate = alloc((size_t)B_ * D_);
    float* h1   = alloc((size_t)B_ * H1_);
    float* h1b  = alloc((size_t)B_ * H1_);
    float* h2   = alloc((size_t)B_ * H2_);
    float* h2v  = alloc((size_t)B_ * H2_);
    float* h2a  = alloc((size_t)B_ * H2_);

    // ---- DAIN ----
    k_dain_avg  <<<B_, D_, 0, stream>>>(src, avg);
    k_dain_mat  <<<B_, D_, 0, stream>>>(avg, dmW, nullptr, m2, 0);
    k_dain_std  <<<B_, D_, 0, stream>>>(src, m2, stdb);
    k_dain_mat  <<<B_, D_, 0, stream>>>(stdb, dsW, nullptr, adb, 1);
    k_dain_mu   <<<B_, D_, 0, stream>>>(src, m2, adb, mub);
    k_dain_mat  <<<B_, D_, 0, stream>>>(mub, dgW, dgb, gate, 2);
    k_dain_write<<<BT_, D_, 0, stream>>>(src, m2, adb, gate, x);

    const dim3 gemmBlk(128);
    // ---- Mamba + gMLP stack ----
    for (int l = 0; l < NL_; ++l) {
        const float* ipW_l  = ipW  + (size_t)l * 2 * DI_ * D_;
        const float* cW_l   = cW   + (size_t)l * DI_ * DC_;
        const float* cb_l   = cb   + (size_t)l * DI_;
        const float* xpW_l  = xpW  + (size_t)l * (DTR_ + 2 * DS_) * DI_;
        const float* dtW_l  = dtW  + (size_t)l * DI_ * DTR_;
        const float* dtb_l  = dtb  + (size_t)l * DI_;
        const float* Alog_l = Alog + (size_t)l * DI_ * DS_;
        const float* Dp_l   = Dp   + (size_t)l * DI_;
        const float* opW_l  = opW  + (size_t)l * D_ * DI_;
        const float* f1W_l  = f1W  + (size_t)l * 2 * DI_ * D_;
        const float* f2W_l  = f2W  + (size_t)l * D_ * DI_;

        // mamba branch
        k_rmsnorm<<<BT_, D_, 0, stream>>>(x, n1w + l * D_, xn);
        k_gemm_wmma<2 * DI_, D_, 0><<<dim3(2 * DI_ / 64, BT_ / 64, 1), gemmBlk, 0, stream>>>(
            xn, ipW_l, xz, nullptr, BT_, D_);
        k_conv_silu<<<BT_, DI_, 0, stream>>>(xz, cW_l, cb_l, xc);
        k_gemm_wmma<DTR_ + 2 * DS_, DI_, 0><<<dim3(1, BT_ / 64, 1), gemmBlk, 0, stream>>>(
            xc, xpW_l, dbl, nullptr, BT_, DI_);
        k_dtproj<<<BT_, DI_, 0, stream>>>(dbl, dtW_l, dtb_l, dtv);
        k_scan  <<<B_, DI_, 0, stream>>>(xc, dtv, dbl, Alog_l, Dp_l, xz, yb);
        k_gemm_wmma<D_, DI_, 1><<<dim3(D_ / 64, BT_ / 64, 1), gemmBlk, 0, stream>>>(
            yb, opW_l, x, nullptr, BT_, DI_);                  // residual add

        // gMLP branch
        k_rmsnorm<<<BT_, D_, 0, stream>>>(x, n2w + l * D_, xn);
        k_gemm_wmma<2 * DI_, D_, 0><<<dim3(2 * DI_ / 64, BT_ / 64, 1), gemmBlk, 0, stream>>>(
            xn, f1W_l, xz, nullptr, BT_, D_);
        k_glu<<<BT_, DI_, 0, stream>>>(xz, xc);
        k_gemm_wmma<D_, DI_, 1><<<dim3(D_ / 64, BT_ / 64, 1), gemmBlk, 0, stream>>>(
            xc, f2W_l, x, nullptr, BT_, DI_);                  // residual add
    }

    // ---- final norm; xn is also the (B, 38400) flat view ----
    k_rmsnorm<<<BT_, D_, 0, stream>>>(x, nfw, xn);

    // ---- heads: split-K WMMA GEMMs (bias pre-filled, atomic accumulate) ----
    const int KC1 = 384;                 // 38400 / 384 = 100 K-slices -> 800 blocks
    const int KC2 = 128;                 // 512 / 128 = 4 K-slices

    // value head
    k_fill_bias<<<B_, 256, 0, stream>>>(h1, vb1, H1_);
    k_gemm_wmma<H1_, FLAT_, 2><<<dim3(H1_ / 64, B_ / 64, FLAT_ / KC1), gemmBlk, 0, stream>>>(
        xn, vW1, h1, nullptr, B_, KC1);
    k_lnrelu<<<B_, 256, 0, stream>>>(h1, vln1w, vln1b, h1b, H1_);
    k_fill_bias<<<B_, 256, 0, stream>>>(h2, vb2, H2_);
    k_gemm_wmma<H2_, H1_, 2><<<dim3(H2_ / 64, B_ / 64, H1_ / KC2), gemmBlk, 0, stream>>>(
        h1b, vW2, h2, nullptr, B_, KC2);
    k_lnrelu<<<B_, 256, 0, stream>>>(h2, vln2w, vln2b, h2v, H2_);

    // advantage head
    k_fill_bias<<<B_, 256, 0, stream>>>(h1, ab1, H1_);
    k_gemm_wmma<H1_, FLAT_, 2><<<dim3(H1_ / 64, B_ / 64, FLAT_ / KC1), gemmBlk, 0, stream>>>(
        xn, aW1, h1, nullptr, B_, KC1);
    k_lnrelu<<<B_, 256, 0, stream>>>(h1, aln1w, aln1b, h1b, H1_);
    k_fill_bias<<<B_, 256, 0, stream>>>(h2, ab2, H2_);
    k_gemm_wmma<H2_, H1_, 2><<<dim3(H2_ / 64, B_ / 64, H1_ / KC2), gemmBlk, 0, stream>>>(
        h1b, aW2, h2, nullptr, B_, KC2);
    k_lnrelu<<<B_, 256, 0, stream>>>(h2, aln2w, aln2b, h2a, H2_);

    // ---- dueling combine + aux ----
    k_head_final<<<1, B_, 0, stream>>>(h2v, h2a, vW3, vb3, aW3, ab3, (float*)d_out);
}
